// point_model_47047071760506
// MI455X (gfx1250) — matrix operand order repaired and checked by reading, not compiled, not run on hardware
//
#include <hip/hip_runtime.h>
#include <hip/hip_bf16.h>

typedef __attribute__((ext_vector_type(16))) _Float16 v16h;
typedef __attribute__((ext_vector_type(8)))  _Float16 v8h;
typedef __attribute__((ext_vector_type(8)))  float    v8f;

// ---------------------------------------------------------------------------
// Pack X (RxK f32, row major) -> Ah (RxKp f16, K zero-padded to mult of 32).
// ---------------------------------------------------------------------------
__global__ __launch_bounds__(256) void pack_a_kernel(
    const float* __restrict__ X, _Float16* __restrict__ Ah, int R, int K, int Kp)
{
    size_t t = (size_t)blockIdx.x * blockDim.x + threadIdx.x;
    if (t >= (size_t)R * Kp) return;
    int k = (int)(t % Kp);
    size_t r = t / Kp;
    Ah[t] = (_Float16)(k < K ? X[r * K + k] : 0.0f);
}

// Pack W (KxC f32) -> Wp (CxKp f16, transposed, zero-padded K).
__global__ __launch_bounds__(256) void pack_w_kernel(
    const float* __restrict__ W, _Float16* __restrict__ Wp, int K, int C, int Kp)
{
    int t = blockIdx.x * blockDim.x + threadIdx.x;
    if (t >= C * Kp) return;
    int k = t % Kp;
    int n = t / Kp;
    Wp[t] = (_Float16)(k < K ? W[(size_t)k * C + n] : 0.0f);
}

// ---------------------------------------------------------------------------
// WMMA GEMM on packed f16 operands: Y(RxC) = Ah(RxKp) @ Wp(CxKp)^T + bias.
// One wave computes a 32x32 output macro-tile (2x2 WMMA tiles, 4 accumulators).
// A fragment (16x32): lane l<16 -> M=l, elems 0..15 = K[k0+16h .. ) contiguous.
// B fragment (32x16): lane holds col n, elems 0..15 = K[k0+16h .. ) contiguous.
// All loads are 16B global_load_b128, branch-free (R%32==0, C%32==0, Kp%32==0).
// ---------------------------------------------------------------------------
__global__ __launch_bounds__(256) void gemm_wmma_packed(
    const _Float16* __restrict__ Ah, const _Float16* __restrict__ Wp,
    const float* __restrict__ bias, float* __restrict__ Y,
    int R, int Kp, int C)
{
    int wave = threadIdx.x >> 5;
    int lane = threadIdx.x & 31;
    int tilesN = C >> 5;
    int tileId = blockIdx.x * 8 + wave;
    int tm = tileId / tilesN;
    int tn = tileId - tm * tilesN;
    if (tm * 32 >= R) return;                 // wave-uniform exit (EXEC all-1 for WMMA)
    int row0 = tm << 5, col0 = tn << 5;
    int half = lane >> 4;
    int l    = lane & 15;

    const _Float16* a0p = Ah + (size_t)(row0 + l) * Kp + half * 16;
    const _Float16* a1p = Ah + (size_t)(row0 + 16 + l) * Kp + half * 16;
    const _Float16* b0p = Wp + (size_t)(col0 + l) * Kp + half * 16;
    const _Float16* b1p = Wp + (size_t)(col0 + 16 + l) * Kp + half * 16;

    v8f acc00 = {}, acc01 = {}, acc10 = {}, acc11 = {};
    for (int k0 = 0; k0 < Kp; k0 += 32) {
        v8h a0lo = *(const v8h*)(a0p + k0);
        v8h a0hi = *(const v8h*)(a0p + k0 + 8);
        v8h a1lo = *(const v8h*)(a1p + k0);
        v8h a1hi = *(const v8h*)(a1p + k0 + 8);
        v8h b0lo = *(const v8h*)(b0p + k0);
        v8h b0hi = *(const v8h*)(b0p + k0 + 8);
        v8h b1lo = *(const v8h*)(b1p + k0);
        v8h b1hi = *(const v8h*)(b1p + k0 + 8);
        __builtin_prefetch((const void*)(a0p + k0 + 32), 0, 1);   // global_prefetch_b8
        __builtin_prefetch((const void*)(a1p + k0 + 32), 0, 1);
        v16h a0, a1, b0, b1;
#pragma unroll
        for (int e = 0; e < 8; ++e) {
            a0[e] = a0lo[e]; a0[e + 8] = a0hi[e];
            a1[e] = a1lo[e]; a1[e + 8] = a1hi[e];
            b0[e] = b0lo[e]; b0[e + 8] = b0hi[e];
            b1[e] = b1lo[e]; b1[e + 8] = b1hi[e];
        }
        acc00 = __builtin_amdgcn_wmma_f32_16x16x32_f16(false, a0, false, b0, (short)0, acc00, false, false);
        acc01 = __builtin_amdgcn_wmma_f32_16x16x32_f16(false, a0, false, b1, (short)0, acc01, false, false);
        acc10 = __builtin_amdgcn_wmma_f32_16x16x32_f16(false, a1, false, b0, (short)0, acc10, false, false);
        acc11 = __builtin_amdgcn_wmma_f32_16x16x32_f16(false, a1, false, b1, (short)0, acc11, false, false);
    }

    float bn0 = bias[col0 + l];
    float bn1 = bias[col0 + 16 + l];
#pragma unroll
    for (int r = 0; r < 8; ++r) {
        int mA = row0 + r + half * 8;
        int mB = row0 + 16 + r + half * 8;
        Y[(size_t)mA * C + col0 + l]      = acc00[r] + bn0;
        Y[(size_t)mA * C + col0 + 16 + l] = acc01[r] + bn1;
        Y[(size_t)mB * C + col0 + l]      = acc10[r] + bn0;
        Y[(size_t)mB * C + col0 + 16 + l] = acc11[r] + bn1;
    }
}

// ---------------------------------------------------------------------------
// BatchNorm (batch statistics over all rows) + ReLU, in place. One block/channel.
// ---------------------------------------------------------------------------
__global__ __launch_bounds__(256) void bn_relu_kernel(
    float* __restrict__ Y, const float* __restrict__ gamma,
    const float* __restrict__ beta, int R, int C)
{
    int c = blockIdx.x;
    int tid = threadIdx.x;
    __shared__ float ssum[256], ssq[256];
    float sum = 0.f, sq = 0.f;
    for (int r = tid; r < R; r += 256) {
        float v = Y[(size_t)r * C + c];
        sum += v; sq += v * v;
    }
    ssum[tid] = sum; ssq[tid] = sq;
    __syncthreads();
    for (int off = 128; off > 0; off >>= 1) {
        if (tid < off) { ssum[tid] += ssum[tid + off]; ssq[tid] += ssq[tid + off]; }
        __syncthreads();
    }
    float mu  = ssum[0] / (float)R;
    float var = ssq[0] / (float)R - mu * mu;
    float sc  = gamma[c] * rsqrtf(var + 1e-5f);
    float sh  = beta[c] - mu * sc;
    for (int r = tid; r < R; r += 256) {
        float v = Y[(size_t)r * C + c] * sc + sh;
        Y[(size_t)r * C + c] = fmaxf(v, 0.f);
    }
}

// --------------------------- data-movement kernels -------------------------
__global__ void transpose_to_pts_kernel(const float* __restrict__ xyz,
                                        float* __restrict__ pts, int B, int N)
{
    int t = blockIdx.x * blockDim.x + threadIdx.x;
    if (t >= B * N) return;
    int b = t / N, n = t - b * N;
    for (int c = 0; c < 3; ++c)
        pts[((size_t)t) * 3 + c] = xyz[((size_t)b * 3 + c) * N + n];
}

__global__ void fps_kernel(const float* __restrict__ pts, int* __restrict__ out_idx,
                           int N, int S)
{
    __shared__ float dist[2048];
    __shared__ float red[256];
    __shared__ int   redi[256];
    __shared__ float fpt[3];
    int b = blockIdx.x;
    int tid = threadIdx.x;
    const float* P = pts + (size_t)b * N * 3;
    for (int i = tid; i < N; i += 256) dist[i] = 1e10f;
    int far = 0;
    for (int s = 0; s < S; ++s) {
        if (tid == 0) {
            out_idx[b * S + s] = far;
            fpt[0] = P[far * 3 + 0]; fpt[1] = P[far * 3 + 1]; fpt[2] = P[far * 3 + 2];
        }
        __syncthreads();
        float best = -1.f; int besti = 0;
        for (int i = tid; i < N; i += 256) {
            float dx = P[i * 3 + 0] - fpt[0];
            float dy = P[i * 3 + 1] - fpt[1];
            float dz = P[i * 3 + 2] - fpt[2];
            float d  = dx * dx + dy * dy + dz * dz;
            float dd = fminf(dist[i], d);
            dist[i] = dd;
            if (dd > best) { best = dd; besti = i; }
        }
        red[tid] = best; redi[tid] = besti;
        __syncthreads();
        for (int off = 128; off > 0; off >>= 1) {
            if (tid < off) {
                if (red[tid + off] > red[tid] ||
                    (red[tid + off] == red[tid] && redi[tid + off] < redi[tid])) {
                    red[tid] = red[tid + off]; redi[tid] = redi[tid + off];
                }
            }
            __syncthreads();
        }
        far = redi[0];
        __syncthreads();
    }
}

__global__ void gather_xyz_kernel(const float* __restrict__ xyz,
                                  const int* __restrict__ idx,
                                  float* __restrict__ out, int N, int S, int B)
{
    int t = blockIdx.x * blockDim.x + threadIdx.x;
    if (t >= B * S) return;
    int b = t / S;
    int pi = idx[t];
    const float* p = xyz + ((size_t)(b * N + pi)) * 3;
    out[(size_t)t * 3 + 0] = p[0];
    out[(size_t)t * 3 + 1] = p[1];
    out[(size_t)t * 3 + 2] = p[2];
}

__global__ void ball_query_kernel(const float* __restrict__ xyz,
                                  const float* __restrict__ nxyz,
                                  int* __restrict__ out,
                                  int N, int S, int ns, float r2, int B)
{
    int t = blockIdx.x * blockDim.x + threadIdx.x;
    if (t >= B * S) return;
    int b = t / S;
    const float* q = nxyz + (size_t)t * 3;
    int* o = out + (size_t)t * ns;
    int cnt = 0, first = -1;
    for (int n = 0; n < N && cnt < ns; ++n) {
        const float* p = xyz + ((size_t)(b * N + n)) * 3;
        float dx = q[0] - p[0], dy = q[1] - p[1], dz = q[2] - p[2];
        if (dx * dx + dy * dy + dz * dz <= r2) {
            if (first < 0) first = n;
            o[cnt++] = n;
        }
    }
    if (first < 0) first = 0;
    for (; cnt < ns; ++cnt) o[cnt] = first;
}

__global__ void group_kernel(const float* __restrict__ xyz,
                             const float* __restrict__ points,
                             const int* __restrict__ idx,
                             const float* __restrict__ nxyz,
                             float* __restrict__ X,
                             int N, int S, int ns, int Cpts, int B)
{
    int t = blockIdx.x * blockDim.x + threadIdx.x;
    if (t >= B * S * ns) return;
    int s = (t / ns) % S;
    int b = t / (ns * S);
    int pi = idx[t];
    if (pi >= N) pi = N - 1;
    const float* p = xyz + ((size_t)(b * N + pi)) * 3;
    const float* q = nxyz + ((size_t)(b * S + s)) * 3;
    int Cin = 3 + Cpts;
    float* o = X + (size_t)t * Cin;
    o[0] = p[0] - q[0]; o[1] = p[1] - q[1]; o[2] = p[2] - q[2];
    const float* f = points + ((size_t)(b * N + pi)) * Cpts;
    for (int c = 0; c < Cpts; ++c) o[3 + c] = f[c];
}

__global__ void groupall_kernel(const float* __restrict__ xyz,
                                const float* __restrict__ points,
                                float* __restrict__ X, int Npts, int Cpts, int B)
{
    int C = 3 + Cpts;
    int t = blockIdx.x * blockDim.x + threadIdx.x;
    if (t >= B * Npts * C) return;
    int c = t % C;
    int r = t / C;
    X[t] = (c < 3) ? xyz[(size_t)r * 3 + c] : points[(size_t)r * Cpts + (c - 3)];
}

__global__ void maxpool_kernel(const float* __restrict__ in, float* __restrict__ out,
                               int G, int ns, int C)
{
    int t = blockIdx.x * blockDim.x + threadIdx.x;
    if (t >= G * C) return;
    int g = t / C, c = t - g * C;
    float m = -3.4e38f;
    for (int j = 0; j < ns; ++j)
        m = fmaxf(m, in[((size_t)g * ns + j) * C + c]);
    out[t] = m;
}

__global__ void bcast_concat_kernel(const float* __restrict__ p1,
                                    const float* __restrict__ pg,
                                    float* __restrict__ X,
                                    int S, int C1, int C2, int B)
{
    int C = C1 + C2;
    int t = blockIdx.x * blockDim.x + threadIdx.x;
    if (t >= B * S * C) return;
    int c = t % C;
    int r = t / C;
    int b = r / S;
    X[t] = (c < C1) ? p1[(size_t)r * C1 + c] : pg[(size_t)b * C2 + (c - C1)];
}

__global__ void knn3_kernel(const float* __restrict__ x1, const float* __restrict__ x2,
                            int* __restrict__ oi, float* __restrict__ ow,
                            int N1, int S2, int B)
{
    int t = blockIdx.x * blockDim.x + threadIdx.x;
    if (t >= B * N1) return;
    int b = t / N1;
    const float* p = x1 + (size_t)t * 3;
    float d0 = 1e30f, d1 = 1e30f, d2 = 1e30f;
    int   i0 = 0, i1 = 0, i2 = 0;
    for (int s = 0; s < S2; ++s) {
        const float* q = x2 + ((size_t)(b * S2 + s)) * 3;
        float dx = p[0] - q[0], dy = p[1] - q[1], dz = p[2] - q[2];
        float d = dx * dx + dy * dy + dz * dz;
        if (d < d0)      { d2 = d1; i2 = i1; d1 = d0; i1 = i0; d0 = d; i0 = s; }
        else if (d < d1) { d2 = d1; i2 = i1; d1 = d;  i1 = s; }
        else if (d < d2) { d2 = d;  i2 = s; }
    }
    float r0 = 1.f / (d0 + 1e-8f), r1 = 1.f / (d1 + 1e-8f), r2 = 1.f / (d2 + 1e-8f);
    float inv = 1.f / (r0 + r1 + r2);
    ow[(size_t)t * 3 + 0] = r0 * inv; ow[(size_t)t * 3 + 1] = r1 * inv; ow[(size_t)t * 3 + 2] = r2 * inv;
    oi[(size_t)t * 3 + 0] = i0;       oi[(size_t)t * 3 + 1] = i1;       oi[(size_t)t * 3 + 2] = i2;
}

__global__ void interp_concat_kernel(const float* __restrict__ p1,
                                     const int* __restrict__ ki,
                                     const float* __restrict__ kw,
                                     const float* __restrict__ p2,
                                     float* __restrict__ X,
                                     int N1, int S2, int C1, int C2, int B)
{
    int C = C1 + C2;
    size_t t = (size_t)blockIdx.x * blockDim.x + threadIdx.x;
    size_t total = (size_t)B * N1 * C;
    if (t >= total) return;
    int c = (int)(t % C);
    size_t r = t / C;
    int b = (int)(r / N1);
    if (c < C1) { X[t] = p1[r * C1 + c]; return; }
    int cc = c - C1;
    float acc = 0.f;
    for (int k = 0; k < 3; ++k) {
        int s = ki[r * 3 + k];
        acc += kw[r * 3 + k] * p2[((size_t)(b * S2 + s)) * C2 + cc];
    }
    X[t] = acc;
}

__global__ void l0cat_kernel(const float* __restrict__ pts, float* __restrict__ out, int total)
{
    int t = blockIdx.x * blockDim.x + threadIdx.x;
    if (t >= total) return;
    int r = t / 6, c = t % 6;
    out[t] = pts[(size_t)r * 3 + (c % 3)];
}

__global__ void transpose_out_kernel(const float* __restrict__ feat,
                                     float* __restrict__ out, int B, int N, int C)
{
    int t = blockIdx.x * blockDim.x + threadIdx.x;
    if (t >= B * N * C) return;
    int n = t % N;
    int c = (t / N) % C;
    int b = t / (N * C);
    out[t] = feat[((size_t)(b * N + n)) * C + c];
}

__global__ void copy_kernel(const float* __restrict__ src, float* __restrict__ dst, int total)
{
    int t = blockIdx.x * blockDim.x + threadIdx.x;
    if (t < total) dst[t] = src[t];
}

// ---------------------------------------------------------------------------
#define LAUNCH(kern, total, ...) \
    kern<<<((total) + 255) / 256, 256, 0, stream>>>(__VA_ARGS__)

extern "C" void kernel_launch(void* const* d_in, const int* in_sizes, int n_in,
                              void* d_out, int out_size, void* d_ws, size_t ws_size,
                              hipStream_t stream)
{
    (void)out_size; (void)ws_size;
    constexpr int B = 16, N0 = 2048;

    // --- map inputs: xyz + 17 layers x (W,b,gamma,beta) ---------------------
    const float* xyz;
    const float* P[17][4];
    if (n_in >= 69 && in_sizes[0] == B * 3 * N0) {
        // insertion order: xyz, sa1, sa2, sa3, fp3, fp2, fp1, conv1
        xyz = (const float*)d_in[0];
        int i = 1;
        for (int l = 0; l < 17; ++l)
            for (int k = 0; k < 4; ++k) P[l][k] = (const float*)d_in[i++];
    } else {
        // jax tree sorted-key order: conv1, fp1(3), fp2(2), fp3(2), sa1..sa3, xyz
        static const int canon[17] = {16, 13, 14, 15, 11, 12, 9, 10, 0, 1, 2, 3, 4, 5, 6, 7, 8};
        int i = 0;
        for (int g = 0; g < 17; ++g)
            for (int k = 0; k < 4; ++k) P[canon[g]][k] = (const float*)d_in[i++];
        xyz = (const float*)d_in[68];
    }

    // --- workspace bump allocator (floats) ----------------------------------
    float* ws = (float*)d_ws;
    size_t o = 0;
    auto A = [&](size_t n) { size_t r = o; o += n; return r; };
    float* pts   = ws + A(98304);      // B*2048*3
    float* l1xyz = ws + A(24576);      // B*512*3
    float* l1pts = ws + A(1048576);    // B*512*128
    float* l2xyz = ws + A(6144);       // B*128*3
    float* l2pts = ws + A(524288);     // B*128*256
    float* l3pts = ws + A(16384);      // B*1024
    float* fp3o  = ws + A(524288);     // B*128*256
    float* fp2o  = ws + A(1048576);    // B*512*128
    float* fp1o  = ws + A(4194304);    // B*2048*128
    float* l0cat = ws + A(196608);     // B*2048*6
    float* knnw  = ws + A(98304);      // B*2048*3
    float* feat  = ws + A(4194304);    // B*2048*128
    int* fpsidx  = (int*)(ws + A(8192));
    int* ball1   = (int*)(ws + A(262144));
    int* ball2   = (int*)(ws + A(131072));
    int* knni    = (int*)(ws + A(98304));
    _Float16* Ah = (_Float16*)(ws + A(10485760));  // 20,971,520 halves (max R*Kp)
    _Float16* Wp = (_Float16*)(ws + A(300000));    // 600,000 halves (max C*Kp)
    float* bufA  = ws + A(33554432);   // 262144*128
    float* bufB  = ws + A(33554432);

    auto gemmL = [&](const float* X, int l, float* Y, int R, int K, int C) {
        int Kp = (K + 31) & ~31;
        LAUNCH(pack_a_kernel, R * Kp, X, Ah, R, K, Kp);
        LAUNCH(pack_w_kernel, C * Kp, P[l][0], Wp, K, C, Kp);
        int tiles = (R / 32) * (C / 32);
        gemm_wmma_packed<<<(tiles + 7) / 8, 256, 0, stream>>>(Ah, Wp, P[l][1], Y, R, Kp, C);
        bn_relu_kernel<<<C, 256, 0, stream>>>(Y, P[l][2], P[l][3], R, C);
    };

    // --- l0 -----------------------------------------------------------------
    LAUNCH(transpose_to_pts_kernel, B * N0, xyz, pts, B, N0);

    // --- SA1: 2048 -> 512, r=0.2, ns=32, [64,64,128] ------------------------
    fps_kernel<<<B, 256, 0, stream>>>(pts, fpsidx, N0, 512);
    LAUNCH(gather_xyz_kernel, B * 512, pts, fpsidx, l1xyz, N0, 512, B);
    LAUNCH(ball_query_kernel, B * 512, pts, l1xyz, ball1, N0, 512, 32, 0.04f, B);
    LAUNCH(group_kernel, B * 512 * 32, pts, pts, ball1, l1xyz, bufA, N0, 512, 32, 3, B);
    gemmL(bufA, 0, bufB, 262144, 6, 64);
    gemmL(bufB, 1, bufA, 262144, 64, 64);
    gemmL(bufA, 2, bufB, 262144, 64, 128);
    LAUNCH(maxpool_kernel, B * 512 * 128, bufB, l1pts, B * 512, 32, 128);

    // --- SA2: 512 -> 128, r=0.4, ns=64, [128,128,256] -----------------------
    fps_kernel<<<B, 256, 0, stream>>>(l1xyz, fpsidx, 512, 128);
    LAUNCH(gather_xyz_kernel, B * 128, l1xyz, fpsidx, l2xyz, 512, 128, B);
    LAUNCH(ball_query_kernel, B * 128, l1xyz, l2xyz, ball2, 512, 128, 64, 0.16f, B);
    LAUNCH(group_kernel, B * 128 * 64, l1xyz, l1pts, ball2, l2xyz, bufA, 512, 128, 64, 128, B);
    gemmL(bufA, 3, bufB, 131072, 131, 128);
    gemmL(bufB, 4, bufA, 131072, 128, 128);
    gemmL(bufA, 5, bufB, 131072, 128, 256);
    LAUNCH(maxpool_kernel, B * 128 * 256, bufB, l2pts, B * 128, 64, 256);

    // --- SA3: group_all, [256,512,1024] -------------------------------------
    LAUNCH(groupall_kernel, B * 128 * 259, l2xyz, l2pts, bufA, 128, 256, B);
    gemmL(bufA, 6, bufB, 2048, 259, 256);
    gemmL(bufB, 7, bufA, 2048, 256, 512);
    gemmL(bufA, 8, bufB, 2048, 512, 1024);
    LAUNCH(maxpool_kernel, B * 1024, bufB, l3pts, B, 128, 1024);

    // --- FP3: broadcast l3 -> l2, concat(256+1024), [256,256] ---------------
    LAUNCH(bcast_concat_kernel, B * 128 * 1280, l2pts, l3pts, bufA, 128, 256, 1024, B);
    gemmL(bufA, 9, bufB, 2048, 1280, 256);
    gemmL(bufB, 10, fp3o, 2048, 256, 256);

    // --- FP2: 3-NN l2 -> l1, concat(128+256), [256,128] ---------------------
    LAUNCH(knn3_kernel, B * 512, l1xyz, l2xyz, knni, knnw, 512, 128, B);
    LAUNCH(interp_concat_kernel, B * 512 * 384, l1pts, knni, knnw, fp3o, bufA, 512, 128, 128, 256, B);
    gemmL(bufA, 11, bufB, 8192, 384, 256);
    gemmL(bufB, 12, fp2o, 8192, 256, 128);

    // --- FP1: 3-NN l1 -> l0, concat(6+128), [128,128,128] -------------------
    LAUNCH(knn3_kernel, B * N0, pts, l1xyz, knni, knnw, N0, 512, B);
    LAUNCH(l0cat_kernel, B * N0 * 6, pts, l0cat, B * N0 * 6);
    LAUNCH(interp_concat_kernel, B * N0 * 134, l0cat, knni, knnw, fp2o, bufA, N0, 512, 6, 128, B);
    gemmL(bufA, 13, bufB, 32768, 134, 128);
    gemmL(bufB, 14, bufA, 32768, 128, 128);
    gemmL(bufA, 15, fp1o, 32768, 128, 128);

    // --- conv1 + outputs ----------------------------------------------------
    gemmL(fp1o, 16, feat, 32768, 128, 128);
    LAUNCH(transpose_out_kernel, B * 128 * N0, feat, (float*)d_out, B, N0, 128);
    LAUNCH(copy_kernel, B * 1024, l3pts, (float*)d_out + (size_t)B * 128 * N0, B * 1024);
}
